// MySelfAttention_68633577390514
// MI455X (gfx1250) — matrix operand order repaired and checked by reading, not compiled, not run on hardware
//
#include <hip/hip_runtime.h>

// Self-attention: B=4, S=2048, DIM=1024, DK=DV=128, fp32 in/out.
// bf16 WMMA (v_wmma_f32_16x16x32_bf16) projections + LDS-staged flash attention.
// Inner loops are split into load-all-fragments then WMMA-chain passes so the
// LDS waits become staggered partial waits instead of per-WMMA full stalls.

#define DIMX 1024
#define DKX  128
#define BX   4
#define SX   2048
#define NROWS (BX * SX)   // 8192

#define KSTR 136          // LDS row stride (elems) for 128-wide K tile (272B, 16B aligned)
#define VSTR 40           // LDS row stride (elems) for 32-wide V^T / W tiles (80B)
#define NIT  (SX / 32)    // 64 key blocks

typedef __bf16 bf16_t;
typedef __attribute__((ext_vector_type(16))) __bf16       v16bf;
typedef __attribute__((ext_vector_type(8)))  __bf16       v8bf;
typedef __attribute__((ext_vector_type(8)))  float        v8f;
typedef __attribute__((ext_vector_type(4)))  float        v4f;
typedef __attribute__((ext_vector_type(4)))  unsigned int v4u;
typedef __attribute__((ext_vector_type(8)))  unsigned int v8u;

static __device__ inline v16bf ld_bf16x16(const bf16_t* p) {
    v8u r = *(const v8u*)p;
    return __builtin_bit_cast(v16bf, r);
}
static __device__ inline v16bf pack2(v4u lo8, v4u hi8) {
    return __builtin_bit_cast(
        v16bf, __builtin_shufflevector(lo8, hi8, 0, 1, 2, 3, 4, 5, 6, 7));
}

static __device__ inline v8f wmma_bf16(v16bf a, v16bf b, v8f c) {
    return __builtin_amdgcn_wmma_f32_16x16x32_bf16(
        false, a, false, b, (short)0, c, false, false);
}

// ---------------------------------------------------------------------------
// Kernel 0: weights fp32 [1024][128] -> bf16 Wt[3][128][1024] (transposed)
// ---------------------------------------------------------------------------
__global__ void wconv_kernel(const float* __restrict__ Wq,
                             const float* __restrict__ Wk,
                             const float* __restrict__ Wv,
                             bf16_t* __restrict__ Wt) {
    int tid = blockIdx.x * blockDim.x + threadIdx.x;
    if (tid >= 3 * DKX * DIMX) return;
    int m = tid / (DKX * DIMX);
    int r = tid % (DKX * DIMX);
    int c = r / DIMX;
    int k = r % DIMX;
    const float* W = (m == 0) ? Wq : ((m == 1) ? Wk : Wv);
    Wt[tid] = (bf16_t)W[k * DKX + c];
}

// ---------------------------------------------------------------------------
// Kernel 1: QKV projection, C[d][s] = W^T x x^T with bf16 WMMA.
// Block = (matrix, 4 adjacent 16-row s-tiles); W k-chunks (8KB) double-buffered
// in LDS and shared by the 4 waves.  All 8 A-fragments loaded before the WMMA
// chain.
// ---------------------------------------------------------------------------
__global__ __launch_bounds__(128) void qkv_proj_kernel(
    const float* __restrict__ x,
    const float* __restrict__ bq, const float* __restrict__ bk,
    const float* __restrict__ bv,
    const bf16_t* __restrict__ Wt,
    bf16_t* __restrict__ q, bf16_t* __restrict__ k, bf16_t* __restrict__ vt)
{
    __shared__ __align__(16) bf16_t lw[2][128 * VSTR];  // 2 x 10KB

    const int tid  = threadIdx.x;
    const int lane = tid & 31;
    const int wid  = tid >> 5;
    const int lo   = lane & 15;
    const int hi   = lane >> 4;

    const int mat   = blockIdx.x >> 7;            // 0..2
    const int grp   = blockIdx.x & 127;
    const int sbase = grp * 64 + wid * 16;        // this wave's 16 rows

    const bf16_t* W    = Wt + (size_t)mat * DKX * DIMX;
    const float*  bias = (mat == 0) ? bq : ((mat == 1) ? bk : bv);

    // prologue: stage k-chunk 0 (thread d=tid copies 32 elems, padded rows)
    {
        const bf16_t* wsrc = W + (size_t)tid * DIMX;
#pragma unroll
        for (int c = 0; c < 4; ++c)
            *(v4u*)&lw[0][tid * VSTR + c * 8] = *(const v4u*)(wsrc + c * 8);
    }
    __syncthreads();

    v8f acc[8] = {};
    const float* xrow = x + (size_t)(sbase + lo) * DIMX;

    for (int kb = 0; kb < DIMX / 32; ++kb) {
        const int cur = kb & 1;
        const int nxt = cur ^ 1;
        const bool hn = (kb + 1) < DIMX / 32;

        // issue next W chunk global loads (no wait yet)
        v4u wrg[4];
        if (hn) {
            const bf16_t* wsrc = W + (size_t)tid * DIMX + (kb + 1) * 32;
#pragma unroll
            for (int c = 0; c < 4; ++c) wrg[c] = *(const v4u*)(wsrc + c * 8);
        }

        // B fragment from x (f32 -> bf16)
        const float* xp = xrow + kb * 32 + hi * 16;
        v4f f0 = *(const v4f*)(xp + 0);
        v4f f1 = *(const v4f*)(xp + 4);
        v4f f2 = *(const v4f*)(xp + 8);
        v4f f3 = *(const v4f*)(xp + 12);
        v16bf bfrag;
#pragma unroll
        for (int i = 0; i < 4; ++i) {
            bfrag[i]      = (bf16_t)f0[i];
            bfrag[4 + i]  = (bf16_t)f1[i];
            bfrag[8 + i]  = (bf16_t)f2[i];
            bfrag[12 + i] = (bf16_t)f3[i];
        }

        // pass 1: load all 8 A-fragments (distinct registers, staggered waits)
        v16bf af[8];
#pragma unroll
        for (int dt = 0; dt < 8; ++dt) {
            const int wa = (dt * 16 + lo) * VSTR + hi * 16;
            af[dt] = pack2(*(const v4u*)&lw[cur][wa],
                           *(const v4u*)&lw[cur][wa + 8]);
        }
        // pass 2: WMMA chain
#pragma unroll
        for (int dt = 0; dt < 8; ++dt)
            acc[dt] = wmma_bf16(af[dt], bfrag, acc[dt]);

        if (hn) {
#pragma unroll
            for (int c = 0; c < 4; ++c)
                *(v4u*)&lw[nxt][tid * VSTR + c * 8] = wrg[c];
        }
        __syncthreads();
    }

    const float scale = 0.08838834764831845f;  // 1/sqrt(128), folded into Q
    const int sg = sbase + lo;
#pragma unroll
    for (int dt = 0; dt < 8; ++dt) {
        const int d0 = dt * 16 + hi * 8;
        v4f b0 = *(const v4f*)(bias + d0);
        v4f b1 = *(const v4f*)(bias + d0 + 4);
        float vals[8];
#pragma unroll
        for (int j = 0; j < 8; ++j) {
            float t = acc[dt][j] + ((j < 4) ? b0[j] : b1[j - 4]);
            if (mat == 0) t *= scale;
            vals[j] = t;
        }
        if (mat < 2) {
            bf16_t* dst = ((mat == 0) ? q : k) + (size_t)sg * DKX + d0;
            v8bf pk;
#pragma unroll
            for (int j = 0; j < 8; ++j) pk[j] = (bf16_t)vals[j];
            *(v4u*)dst = __builtin_bit_cast(v4u, pk);
        } else {
            const int b  = sg >> 11;
            const int sl = sg & (SX - 1);
#pragma unroll
            for (int j = 0; j < 8; ++j)
                vt[((size_t)(b * DKX + d0 + j)) * SX + sl] = (bf16_t)vals[j];
        }
    }
}

// ---------------------------------------------------------------------------
// Kernel 2: flash attention.  Block = 4 waves = 4 adjacent q-tiles (64 rows,
// one batch).  K (32x128) / V^T (128x32) tiles double-buffered in LDS, shared
// by the 4 waves.  Fragments loaded in batches ahead of each WMMA chain.
// ---------------------------------------------------------------------------
__global__ __launch_bounds__(128) void flash_attn_kernel(
    const bf16_t* __restrict__ q, const bf16_t* __restrict__ k,
    const bf16_t* __restrict__ vt, float* __restrict__ out)
{
    __shared__ __align__(16) bf16_t lk[2][32 * KSTR];   // 2 x 8.5KB
    __shared__ __align__(16) bf16_t lv[2][128 * VSTR];  // 2 x 10KB

    const int tid  = threadIdx.x;
    const int lane = tid & 31;
    const int wid  = tid >> 5;
    const int lo   = lane & 15;
    const int hi   = lane >> 4;

    const int sgbase = (blockIdx.x * 4 + wid) * 16;  // this wave's query rows
    const int b      = blockIdx.x >> 5;              // batch (32 blocks/batch)

    const bf16_t* kbase = k  + (size_t)b * SX * DKX;
    const bf16_t* vbase = vt + (size_t)b * DKX * SX;

    // staging roles: K -> thread (row, quarter); V -> thread d = tid
    const int kr_row = tid >> 2;
    const int kr_q   = (tid & 3) * 32;

    // preload Q as 4 B-fragments (Q^T: K-dim=d, cols=s)
    v16bf qf[4];
    const bf16_t* qrow = q + (size_t)(sgbase + lo) * DKX + hi * 8;
#pragma unroll
    for (int db = 0; db < 4; ++db) qf[db] = pack2(*(const v4u*)(qrow + db * 32), *(const v4u*)(qrow + db * 32 + 16));

    v8f   ot[8] = {};
    float m = -3.0e38f, l = 0.0f;

    // prologue: stage key block 0 into buffer 0
    {
        const bf16_t* ks = kbase + (size_t)kr_row * DKX + kr_q;
        const bf16_t* vs = vbase + (size_t)tid * SX;
#pragma unroll
        for (int c = 0; c < 4; ++c) {
            *(v4u*)&lk[0][kr_row * KSTR + kr_q + c * 8] = *(const v4u*)(ks + c * 8);
            *(v4u*)&lv[0][tid * VSTR + c * 8]           = *(const v4u*)(vs + c * 8);
        }
    }
    __syncthreads();

    for (int it = 0; it < NIT; ++it) {
        const int  cur = it & 1;
        const int  nxt = cur ^ 1;
        const int  tb  = it * 32;
        const bool hn  = (it + 1) < NIT;

        // issue next block's global loads (overlap with compute)
        v4u krg[4], vrg[4];
        if (hn) {
            const bf16_t* ks = kbase + (size_t)(tb + 32 + kr_row) * DKX + kr_q;
            const bf16_t* vs = vbase + (size_t)tid * SX + tb + 32;
#pragma unroll
            for (int c = 0; c < 4; ++c) {
                krg[c] = *(const v4u*)(ks + c * 8);
                vrg[c] = *(const v4u*)(vs + c * 8);
            }
        }

        // ---- S^T tiles: load all 8 K fragments, then 8-WMMA chain ----
        v8f s0 = {}, s1 = {};
        const int ka0 = lo * KSTR + hi * 8;
        const int ka1 = (16 + lo) * KSTR + hi * 8;
        v16bf kf0[4], kf1[4];
#pragma unroll
        for (int db = 0; db < 4; ++db) {
            const int o = db * 32;
            kf0[db] = pack2(*(const v4u*)&lk[cur][ka0 + o],
                            *(const v4u*)&lk[cur][ka0 + o + 16]);
            kf1[db] = pack2(*(const v4u*)&lk[cur][ka1 + o],
                            *(const v4u*)&lk[cur][ka1 + o + 16]);
        }
#pragma unroll
        for (int db = 0; db < 4; ++db) {
            s0 = wmma_bf16(kf0[db], qf[db], s0);
            s1 = wmma_bf16(kf1[db], qf[db], s1);
        }

        // ---- online softmax (per s-column; column split over lane pair) ----
        float mb = fmaxf(s0[0], s1[0]);
#pragma unroll
        for (int j = 1; j < 8; ++j) mb = fmaxf(mb, fmaxf(s0[j], s1[j]));
        mb = fmaxf(mb, __shfl_xor(mb, 16));
        const float mn    = fmaxf(m, mb);
        const float alpha = __expf(m - mn);
        float p0[8], p1[8], rs = 0.0f;
#pragma unroll
        for (int j = 0; j < 8; ++j) {
            p0[j] = __expf(s0[j] - mn);
            p1[j] = __expf(s1[j] - mn);
            rs += p0[j] + p1[j];
        }
        rs += __shfl_xor(rs, 16);
        l = l * alpha + rs;
        m = mn;

        // ---- P^T B-fragment (32t x 16s) via half-wave exchange ----
        v16bf pb;
#pragma unroll
        for (int j = 0; j < 8; ++j) {
            float send  = hi ? p0[j] : p1[j];
            float recv  = __shfl_xor(send, 16);
            float first = hi ? recv : p0[j];
            float secnd = hi ? p1[j] : recv;
            pb[j]     = (bf16_t)first;
            pb[8 + j] = (bf16_t)secnd;
        }

        // ---- O^T = alpha*O^T + V^T x P^T over 8 d-tiles, in groups of 4 ----
#pragma unroll
        for (int g = 0; g < 2; ++g) {
            v16bf vf[4];
#pragma unroll
            for (int d2 = 0; d2 < 4; ++d2) {
                const int va = ((g * 4 + d2) * 16 + lo) * VSTR + hi * 16;
                vf[d2] = pack2(*(const v4u*)&lv[cur][va],
                               *(const v4u*)&lv[cur][va + 8]);
            }
#pragma unroll
            for (int d2 = 0; d2 < 4; ++d2) {
                const int dt = g * 4 + d2;
#pragma unroll
                for (int j = 0; j < 8; ++j) ot[dt][j] *= alpha;
                ot[dt] = wmma_bf16(vf[d2], pb, ot[dt]);
            }
        }

        // ---- commit next block to the other LDS buffer ----
        if (hn) {
#pragma unroll
            for (int c = 0; c < 4; ++c) {
                *(v4u*)&lk[nxt][kr_row * KSTR + kr_q + c * 8] = krg[c];
                *(v4u*)&lv[nxt][tid * VSTR + c * 8]           = vrg[c];
            }
        }
        __syncthreads();
    }

    // ---- normalize and store out[b][s][d]: 8 consecutive d per lane ----
    const float rl = 1.0f / l;
    float* orow = out + (size_t)(sgbase + lo) * DKX;
#pragma unroll
    for (int dt = 0; dt < 8; ++dt) {
        const int d0 = dt * 16 + hi * 8;
        v4f o0, o1;
#pragma unroll
        for (int j = 0; j < 4; ++j) {
            o0[j] = ot[dt][j] * rl;
            o1[j] = ot[dt][4 + j] * rl;
        }
        *(v4f*)(orow + d0)     = o0;
        *(v4f*)(orow + d0 + 4) = o1;
    }
}

// ---------------------------------------------------------------------------
extern "C" void kernel_launch(void* const* d_in, const int* in_sizes, int n_in,
                              void* d_out, int out_size, void* d_ws,
                              size_t ws_size, hipStream_t stream) {
    const float* x  = (const float*)d_in[0];
    const float* Wq = (const float*)d_in[1];
    const float* bq = (const float*)d_in[2];
    const float* Wk = (const float*)d_in[3];
    const float* bk = (const float*)d_in[4];
    const float* Wv = (const float*)d_in[5];
    const float* bv = (const float*)d_in[6];
    float* out = (float*)d_out;

    // workspace (bf16 elems): Wt | q | k | vt  (~6.75 MB)
    bf16_t* Wt = (bf16_t*)d_ws;                 // [3][128][1024]
    bf16_t* q  = Wt + (size_t)3 * DKX * DIMX;   // [8192][128] (pre-scaled)
    bf16_t* k  = q + (size_t)NROWS * DKX;       // [8192][128]
    bf16_t* vt = k + (size_t)NROWS * DKX;       // [4][128][2048]

    {
        int n = 3 * DKX * DIMX;
        wconv_kernel<<<(n + 255) / 256, 256, 0, stream>>>(Wq, Wk, Wv, Wt);
    }
    // 3 matrices x 128 groups of 4 s-tiles
    qkv_proj_kernel<<<384, 128, 0, stream>>>(x, bq, bk, bv, Wt, q, k, vt);
    // 128 blocks x 4 q-tiles
    flash_attn_kernel<<<128, 128, 0, stream>>>(q, k, vt, out);
}
